// GCNCCT_54254026883726
// MI455X (gfx1250) — compile-verified, matrix operand order checked by reading, not run on hardware
//
#include <hip/hip_runtime.h>

// ---------------- problem constants ----------------
#define Bc     16
#define Tc     360
#define Vc     25
#define Nc     9000          // T*V
#define Ec     72000
#define Hc     128
#define Mc     (Bc*Nc)       // 144000 rows
#define NCLSc  12
#define NAUXc  8
#define EPSc   1e-5f
#define INV_M  (1.0f/(float)Mc)
#define INV_N  (1.0f/(float)Nc)

typedef __attribute__((ext_vector_type(16))) _Float16 v16h;
typedef __attribute__((ext_vector_type(8)))  float    v8f;

union HF16 { v16h v; uint4 q[2]; };

// cyclic time shift of a node id: (t,v) -> ((t+sT)%T, v)
__device__ __forceinline__ int shift_node(int n, int sT) {
    if (sT == 0) return n;
    int t = n / Vc;
    int v = n - t * Vc;
    t += sT; if (t >= Tc) t -= Tc;
    return t * Vc + v;
}

// ---------------- setup: degrees & symmetric-norm coefficients ----------------
__global__ void k_fill(float* __restrict__ p, float v, int n) {
    int i = blockIdx.x * 256 + threadIdx.x;
    if (i < n) p[i] = v;
}
__global__ void k_deg(const int* __restrict__ dst, float* __restrict__ deg) {
    int e = blockIdx.x * 256 + threadIdx.x;
    if (e < Ec) atomicAdd(&deg[dst[e]], 1.0f);
}
__global__ void k_rsqrt_ip(float* __restrict__ p, int n) {
    int i = blockIdx.x * 256 + threadIdx.x;
    if (i < n) p[i] = rsqrtf(p[i]);
}
__global__ void k_coef(const int* __restrict__ src, const int* __restrict__ dst,
                       const float* __restrict__ dinv, float* __restrict__ coef) {
    int e = blockIdx.x * 256 + threadIdx.x;
    if (e < Ec) coef[e] = dinv[src[e]] * dinv[dst[e]];
}
__global__ void k_zero(float* __restrict__ p, int n) {
    int i = blockIdx.x * 256 + threadIdx.x;
    if (i < n) p[i] = 0.0f;
}

// ---------------- layer 0 GEMM (K = DIN = 2, trivial) ----------------
__global__ void k_in_gemm(const float* __restrict__ x, const float* __restrict__ Win,
                          float* __restrict__ Y) {
    int idx = blockIdx.x * 256 + threadIdx.x;     // Mc*Hc total, exact grid
    int row = idx >> 7, ch = idx & 127;
    Y[idx] = x[row * 2] * Win[ch] + x[row * 2 + 1] * Win[Hc + ch];
}

// ---------------- W (f32 [k][n]) -> transposed f16 [n][k] ----------------
__global__ void k_convert_wt(const float* __restrict__ W, _Float16* __restrict__ Wt) {
    int idx = blockIdx.x * 256 + threadIdx.x;     // Hc*Hc total
    int k = idx >> 7, n = idx & 127;
    Wt[n * Hc + k] = (_Float16)W[idx];
}

// ---------------- WMMA GEMM: Y[M,128] = A_f16[M,128] x W[128,128] ----------------
// Block = 256 thr = 8 waves; each block does 128 rows x 128 cols.
// Wave w: 16-row stripe, 8 col-tiles, K-loop of 4 x 32 -> 32 v_wmma per wave.
__global__ __launch_bounds__(256) void k_wmma_gemm(const _Float16* __restrict__ A,
                                                   const _Float16* __restrict__ Wt, // [n][k]
                                                   float* __restrict__ Y) {
    __shared__ __align__(16) _Float16 sWt[Hc * Hc];   // 32 KB
    int tid = threadIdx.x;
    {   // cooperative 32KB copy of transposed weights into LDS
        const uint4* g = (const uint4*)Wt;
        uint4* s = (uint4*)sWt;
        #pragma unroll
        for (int i = 0; i < 8; ++i) s[tid + i * 256] = g[tid + i * 256];
    }
    __syncthreads();

    int wave = tid >> 5;
    int lane = tid & 31;
    int lo16 = (lane < 16);
    int l15  = lane & 15;
    int rowBase = blockIdx.x * 128 + wave * 16;

    // A fragment rows: lanes 0-15 and 16-31 both hold M = l15, different K halves.
    const _Float16* arow = A + (size_t)(rowBase + l15) * Hc;
    int kbase = lo16 ? 0 : 8;

    v8f acc[8] = {};
    #pragma unroll
    for (int k0 = 0; k0 < Hc; k0 += 32) {
        HF16 a;
        a.q[0] = *(const uint4*)(arow + k0 + kbase);        // K = k0+kbase .. +7
        a.q[1] = *(const uint4*)(arow + k0 + kbase + 16);   // K = k0+kbase+16 .. +23
        #pragma unroll
        for (int nt = 0; nt < 8; ++nt) {
            int col = nt * 16 + l15;
            HF16 b;
            b.q[0] = *(const uint4*)(&sWt[col * Hc + k0 + kbase]);
            b.q[1] = *(const uint4*)(&sWt[col * Hc + k0 + kbase + 16]);
            acc[nt] = __builtin_amdgcn_wmma_f32_16x16x32_f16(
                false, a.v, false, b.v, (short)0, acc[nt], false, false);
        }
    }
    // D layout: VGPR i, lanes 0-15 -> M=i, lanes 16-31 -> M=8+i; N = l15
    #pragma unroll
    for (int nt = 0; nt < 8; ++nt) {
        int col = nt * 16 + l15;
        #pragma unroll
        for (int i = 0; i < 8; ++i) {
            int r = rowBase + i + (lo16 ? 0 : 8);
            Y[(size_t)r * Hc + col] = acc[nt][i];
        }
    }
}

// ---------------- aggregation: AGG = bias + selfcoef*Y[sigma(n)] + scatter ----------------
__global__ void k_agg_init(const float* __restrict__ Y, float* __restrict__ AGG,
                           const float* __restrict__ dinv, const float* __restrict__ bias,
                           int shiftT) {
    int idx = blockIdx.x * 256 + threadIdx.x;     // Mc*Hc, exact
    int row = idx >> 7, ch = idx & 127;
    int n = row % Nc;
    int sn = shift_node(n, shiftT);
    float sc = dinv[n] * dinv[n];
    AGG[idx] = bias[ch] + sc * Y[(size_t)(row - n + sn) * Hc + ch];
}

__global__ void k_agg_edges(const float* __restrict__ Y, float* __restrict__ AGG,
                            const int* __restrict__ src, const int* __restrict__ dst,
                            const float* __restrict__ coef, int shiftT) {
    int e = blockIdx.x;                // edge
    int b = blockIdx.y;                // batch
    int s = shift_node(src[e], shiftT);
    int d = dst[e];
    float c = coef[e];
    const float* y = Y   + (size_t)(b * Nc + s) * Hc;
    float*       a = AGG + (size_t)(b * Nc + d) * Hc;
    atomicAdd(a + threadIdx.x, c * y[threadIdx.x]);   // L2-resident atomic
}

// ---------------- BatchNorm stats: per-channel sum / sumsq over B*N ----------------
__global__ void k_bn_stats(const float* __restrict__ AGG, float* __restrict__ stats) {
    int tid = threadIdx.x;
    int ch = tid & 127;
    float s = 0.f, s2 = 0.f;
    // stride is a multiple of 128 -> channel is invariant per thread
    for (size_t i = (size_t)blockIdx.x * 256 + tid; i < (size_t)Mc * Hc; i += (size_t)512 * 256) {
        float v = AGG[i];
        s += v; s2 += v * v;
    }
    __shared__ float ls[256], ls2[256];
    ls[tid] = s; ls2[tid] = s2;
    __syncthreads();
    if (tid < 128) {
        atomicAdd(&stats[ch],       ls[tid]  + ls[tid + 128]);
        atomicAdd(&stats[128 + ch], ls2[tid] + ls2[tid + 128]);
    }
}

// ---------------- BN + ReLU -> f16 features for next WMMA GEMM ----------------
__global__ void k_bn_relu_f16(const float* __restrict__ AGG, const float* __restrict__ stats,
                              const float* __restrict__ g, const float* __restrict__ be,
                              _Float16* __restrict__ Hf) {
    int idx = blockIdx.x * 256 + threadIdx.x;
    int ch = idx & 127;
    float mean = stats[ch] * INV_M;
    float var  = stats[128 + ch] * INV_M - mean * mean;
    float rstd = rsqrtf(var + EPSc);
    float v = g[ch] * ((AGG[idx] - mean) * rstd) + be[ch];
    Hf[idx] = (_Float16)fmaxf(v, 0.0f);
}

// ---------------- BN + ReLU + global-mean-pool accumulate ----------------
__global__ void k_bn_relu_pool(const float* __restrict__ AGG, const float* __restrict__ stats,
                               const float* __restrict__ g, const float* __restrict__ be,
                               float* __restrict__ pooled) {
    int b = blockIdx.y;
    int tid = threadIdx.x;
    int ch = tid & 127;
    int half = tid >> 7;
    float mean = stats[ch] * INV_M;
    float rstd = rsqrtf(stats[128 + ch] * INV_M - mean * mean + EPSc);
    float gg = g[ch], bb = be[ch];
    float sum = 0.f;
    int r0 = blockIdx.x * 250;                    // 36 chunks * 250 = N
    for (int r = r0 + half; r < r0 + 250; r += 2) {
        float v = gg * ((AGG[(size_t)(b * Nc + r) * Hc + ch] - mean) * rstd) + bb;
        sum += fmaxf(v, 0.0f);
    }
    __shared__ float ls[256];
    ls[tid] = sum;
    __syncthreads();
    if (tid < 128) atomicAdd(&pooled[b * Hc + ch], ls[tid] + ls[tid + 128]);
}

// ---------------- head: pooled@Wf + bf, log_softmax -> output slice ----------------
__global__ void k_head_final(const float* __restrict__ pooled, const float* __restrict__ Wf,
                             const float* __restrict__ bf, float* __restrict__ outp,
                             int rowStride) {
    __shared__ float lg[Bc][NCLSc];
    int tid = threadIdx.x;
    if (tid < Bc * NCLSc) {
        int b = tid / NCLSc, c = tid % NCLSc;
        float acc = bf[c];
        const float* p = pooled + b * Hc;
        #pragma unroll 4
        for (int k = 0; k < Hc; ++k) acc += (p[k] * INV_N) * Wf[k * NCLSc + c];
        lg[b][c] = acc;
    }
    __syncthreads();
    if (tid < Bc * NCLSc) {
        int b = tid / NCLSc, c = tid % NCLSc;
        float m = -1e30f;
        for (int j = 0; j < NCLSc; ++j) m = fmaxf(m, lg[b][j]);
        float s = 0.f;
        for (int j = 0; j < NCLSc; ++j) s += expf(lg[b][j] - m);
        outp[b * rowStride + c] = lg[b][c] - m - logf(s);
    }
}

// ---------------- host orchestration ----------------
extern "C" void kernel_launch(void* const* d_in, const int* in_sizes, int n_in,
                              void* d_out, int out_size, void* d_ws, size_t ws_size,
                              hipStream_t stream) {
    const float* x        = (const float*)d_in[0];
    const int*   ei       = (const int*)  d_in[1];
    const float* W_in     = (const float*)d_in[2];
    const float* b_in     = (const float*)d_in[3];
    const float* g_in     = (const float*)d_in[4];
    const float* be_in    = (const float*)d_in[5];
    const float* shared_W = (const float*)d_in[6];
    const float* shared_b = (const float*)d_in[7];
    const float* shared_g = (const float*)d_in[8];
    const float* shared_be= (const float*)d_in[9];
    const float* main_Wg  = (const float*)d_in[10];
    const float* main_bg  = (const float*)d_in[11];
    const float* main_g   = (const float*)d_in[12];
    const float* main_be  = (const float*)d_in[13];
    const float* main_Wf  = (const float*)d_in[14];
    const float* main_bf  = (const float*)d_in[15];
    const float* aux_Wg   = (const float*)d_in[16];
    const float* aux_bg   = (const float*)d_in[17];
    const float* aux_g    = (const float*)d_in[18];
    const float* aux_be   = (const float*)d_in[19];
    const float* aux_Wf   = (const float*)d_in[20];
    const float* aux_bf   = (const float*)d_in[21];

    const int* srcI = ei;
    const int* dstI = ei + Ec;

    // workspace partition (~185 MB; Y+AGG both fit MI455X's 192MB L2)
    char* w = (char*)d_ws;
    float*    Y    = (float*)w;     w += (size_t)Mc * Hc * 4;   // gemm out
    float*    AGG  = (float*)w;     w += (size_t)Mc * Hc * 4;   // aggregated
    _Float16* Hf   = (_Float16*)w;  w += (size_t)Mc * Hc * 2;   // f16 features
    _Float16* Wt   = (_Float16*)w;  w += (size_t)Hc * Hc * 2;   // f16 W^T
    float*    coef = (float*)w;     w += (size_t)Ec * 4;
    float*    dinv = (float*)w;     w += (size_t)Nc * 4;
    float*    stats= (float*)w;     w += 256 * 4;               // [sum|sumsq] per ch
    float*    pooled=(float*)w;     w += (size_t)Bc * Hc * 4;

    // ---- graph normalization coefficients ----
    k_fill <<<(Nc + 255) / 256, 256, 0, stream>>>(dinv, 1.0f, Nc);   // self-loop deg
    k_deg  <<<(Ec + 255) / 256, 256, 0, stream>>>(dstI, dinv);
    k_rsqrt_ip<<<(Nc + 255) / 256, 256, 0, stream>>>(dinv, Nc);
    k_coef <<<(Ec + 255) / 256, 256, 0, stream>>>(srcI, dstI, dinv, coef);

    auto agg_bn = [&](const float* bias, int shiftT) {
        k_agg_init <<<Mc * Hc / 256, 256, 0, stream>>>(Y, AGG, dinv, bias, shiftT);
        k_agg_edges<<<dim3(Ec, Bc), 128, 0, stream>>>(Y, AGG, srcI, dstI, coef, shiftT);
        k_zero     <<<1, 256, 0, stream>>>(stats, 256);
        k_bn_stats <<<512, 256, 0, stream>>>(AGG, stats);
    };

    // ---- input layer (DIN=2, plain GEMM) ----
    k_in_gemm<<<Mc * Hc / 256, 256, 0, stream>>>(x, W_in, Y);
    agg_bn(b_in, 0);
    k_bn_relu_f16<<<Mc * Hc / 256, 256, 0, stream>>>(AGG, stats, g_in, be_in, Hf);

    // ---- two shared 128->128 layers (WMMA) ----
    for (int i = 0; i < 2; ++i) {
        k_convert_wt<<<Hc * Hc / 256, 256, 0, stream>>>(shared_W + (size_t)i * Hc * Hc, Wt);
        k_wmma_gemm <<<Mc / 128, 256, 0, stream>>>(Hf, Wt, Y);
        agg_bn(shared_b + i * Hc, 0);
        k_bn_relu_f16<<<Mc * Hc / 256, 256, 0, stream>>>(AGG, stats,
                                                         shared_g + i * Hc, shared_be + i * Hc, Hf);
    }

    // ---- heads: GCN(WMMA) + BN+ReLU+pool + linear + log_softmax ----
    float* out = (float*)d_out;
    auto head = [&](const float* Wg, const float* bg, const float* g, const float* be,
                    const float* Wf, const float* bf, int shiftT, float* outp, int rowStride) {
        k_convert_wt<<<Hc * Hc / 256, 256, 0, stream>>>(Wg, Wt);
        // (P.h).W == P.(h.W): shift applied to source indices inside aggregation
        k_wmma_gemm <<<Mc / 128, 256, 0, stream>>>(Hf, Wt, Y);
        agg_bn(bg, shiftT);
        k_zero        <<<8, 256, 0, stream>>>(pooled, Bc * Hc);
        k_bn_relu_pool<<<dim3(36, Bc), 256, 0, stream>>>(AGG, stats, g, be, pooled);
        k_head_final  <<<1, 192, 0, stream>>>(pooled, Wf, bf, outp, rowStride);
    };

    head(main_Wg, main_bg, main_g, main_be, main_Wf, main_bf, 0, out, NCLSc);

    const int wsT = Tc / (NAUXc + 1);   // 40
    for (int i = 0; i < NAUXc; ++i) {
        head(aux_Wg + (size_t)i * Hc * Hc, aux_bg + i * Hc, aux_g + i * Hc, aux_be + i * Hc,
             aux_Wf + (size_t)i * Hc * NCLSc, aux_bf + i * NCLSc,
             wsT * (i + 1),
             out + Bc * NCLSc + i * NCLSc,    // o_aux[b, i, c] flat
             NAUXc * NCLSc);
    }
}
